// AspectOpinionJointClassifier_66202625900974
// MI455X (gfx1250) — matrix operand order repaired and checked by reading, not compiled, not run on hardware
//
#include <hip/hip_runtime.h>

typedef __bf16 v16bf __attribute__((ext_vector_type(16)));
typedef __bf16 v8bf  __attribute__((ext_vector_type(8)));
typedef float  v8f   __attribute__((ext_vector_type(8)));
typedef unsigned int u32x4 __attribute__((ext_vector_type(4)));
typedef int i32x4 __attribute__((ext_vector_type(4)));
typedef int i32x8 __attribute__((ext_vector_type(8)));

#define WMMA_BF16(a, b, c) \
  __builtin_amdgcn_wmma_f32_16x16x32_bf16(false, (a), false, (b), (short)0, (c), false, false)

__device__ __forceinline__ v16bf mk16(v8bf lo, v8bf hi) {
  v16bf r;
#pragma unroll
  for (int i = 0; i < 8; ++i) { r[i] = lo[i]; r[i + 8] = hi[i]; }
  return r;
}
__device__ __forceinline__ v8bf ld8(const __bf16* p) {
  return *reinterpret_cast<const v8bf*>(p);
}
__device__ __forceinline__ float geluf(float x) {
  return 0.5f * x * (1.0f + erff(x * 0.70710678118654752f));
}

// ---- Tensor Data Mover: 2D tile (dim0 x dim1, 2-byte elems) global -> LDS -------------
// pad_bits packs D# group1 bits [20] pad_enable, [24:22] pad_interval, [31:25] pad_amount.
__device__ __forceinline__ void tdm_load_2d(unsigned int lds_addr, const void* gaddr,
                                            unsigned int dim0, unsigned int dim1,
                                            unsigned int stride0_elems,
                                            unsigned int pad_bits) {
  unsigned long long ga = (unsigned long long)(size_t)gaddr;
  u32x4 g0;
  g0[0] = 1u;                                                   // count=1, user descriptor
  g0[1] = lds_addr;                                             // LDS byte address
  g0[2] = (unsigned int)(ga & 0xffffffffu);                     // global_addr[31:0]
  g0[3] = (unsigned int)((ga >> 32) & 0x01ffffffu) | (2u << 30);// global_addr[56:32], type=2
  i32x8 g1;
  g1[0] = (int)((1u << 16) | pad_bits);        // data_size=1 (2 bytes) + pad controls
  g1[1] = (int)(dim0 << 16);                   // tensor_dim0[15:0] (barrier addr = 0)
  g1[2] = (int)((dim0 >> 16) | (dim1 << 16));  // tensor_dim0[31:16] | tensor_dim1[15:0]
  g1[3] = (int)((dim1 >> 16) | (dim0 << 16));  // tensor_dim1[31:16] | tile_dim0
  g1[4] = (int)dim1;                           // tile_dim1 (tile_dim2 = 0)
  g1[5] = (int)stride0_elems;                  // tensor_dim0_stride[31:0]
  g1[6] = 0;
  g1[7] = 0;
  i32x4 z4 = {0, 0, 0, 0};
#if defined(__clang_major__) && (__clang_major__ >= 23)
  i32x8 z8 = {0, 0, 0, 0, 0, 0, 0, 0};
  __builtin_amdgcn_tensor_load_to_lds(g0, g1, z4, z4, z8, 0);
#else
  __builtin_amdgcn_tensor_load_to_lds(g0, g1, z4, z4, 0);
#endif
}

// ---------------- f32 -> bf16 conversion ----------------
__global__ void k_cvt_bf16(const float* __restrict__ in, __bf16* __restrict__ out, int n) {
  int i = blockIdx.x * blockDim.x + threadIdx.x;
  int stride = gridDim.x * blockDim.x;
  for (; i < n; i += stride) out[i] = (__bf16)in[i];
}

// ---------------- token weights: softmax over S of logits[:,:,1:5], summed over classes ----
__global__ void k_tok_weights(const float* __restrict__ alog, const float* __restrict__ olog,
                              float* __restrict__ aw, float* __restrict__ ow) {
  const int S = 1024, T = 5;
  int b = blockIdx.x, tid = threadIdx.x;
  __shared__ float red[256];
  float acc[2][4] = {};
  for (int br = 0; br < 2; ++br) {
    const float* L = br ? olog : alog;
    for (int t = 1; t < 5; ++t) {
      float v[4]; float mx = -3.0e38f;
#pragma unroll
      for (int i = 0; i < 4; ++i) {
        int s = tid + i * 256;
        v[i] = L[(b * S + s) * T + t];
        mx = fmaxf(mx, v[i]);
      }
      red[tid] = mx; __syncthreads();
      for (int o = 128; o > 0; o >>= 1) { if (tid < o) red[tid] = fmaxf(red[tid], red[tid + o]); __syncthreads(); }
      mx = red[0]; __syncthreads();
      float e[4]; float sm = 0.f;
#pragma unroll
      for (int i = 0; i < 4; ++i) { e[i] = expf(v[i] - mx); sm += e[i]; }
      red[tid] = sm; __syncthreads();
      for (int o = 128; o > 0; o >>= 1) { if (tid < o) red[tid] += red[tid + o]; __syncthreads(); }
      float inv = 1.0f / red[0]; __syncthreads();
#pragma unroll
      for (int i = 0; i < 4; ++i) acc[br][i] += e[i] * inv;
    }
  }
#pragma unroll
  for (int i = 0; i < 4; ++i) {
    int s = tid + i * 256;
    aw[b * S + s] = acc[0][i];
    ow[b * S + s] = acc[1][i];
  }
}

// ---------------- WMMA GEMM: C[M,N](bf16) = A[M,K](bf16) @ W[K,N](bf16) + bias[N] --------
// A tile staged by the Tensor Data Mover with LDS row padding (stride 40 bf16).
__global__ __launch_bounds__(256) void k_gemm_bf16(
    const __bf16* __restrict__ A, const __bf16* __restrict__ W,
    const float* __restrict__ bias, __bf16* __restrict__ C,
    int M, int N, int K) {
  (void)M;
  __shared__ __align__(16) __bf16 As[128 * 40 + 8];  // +8: TDM pads after last row too
  __shared__ __align__(16) __bf16 Wt[64 * 40];
  const int tid = threadIdx.x;
  const int lane = tid & 31, wave = tid >> 5;
  const int wm = wave & 3, wn = wave >> 2;
  const int m0 = blockIdx.y * 128, n0 = blockIdx.x * 64;
  const int ln = lane & 15, half = lane >> 4;
  // pad_interval=3 (16 DWORDs = one 64B row), pad_amount=3 (4 DWORDs = 16B -> stride 40)
  const unsigned int padA = (1u << 20) | (3u << 22) | (3u << 25);

  v8f acc[2][2];
#pragma unroll
  for (int i = 0; i < 2; ++i)
#pragma unroll
    for (int j = 0; j < 2; ++j)
#pragma unroll
      for (int e = 0; e < 8; ++e) acc[i][j][e] = 0.0f;

  for (int k0 = 0; k0 < K; k0 += 32) {
    // A tile 128x32 via TDM (wave 0 issues; TENSORcnt-tracked)
    if (wave == 0) {
      tdm_load_2d((unsigned int)(size_t)As, A + (size_t)m0 * K + k0,
                  /*dim0=*/32, /*dim1=*/128, /*stride0=*/(unsigned int)K, padA);
    }
    // W tile 32x64 transposed into LDS -> Wt[n][k]
    {
      int r = tid >> 3, cc = (tid & 7) * 8;
      if (k0 + 32 < K) __builtin_prefetch(&W[(size_t)(k0 + 32 + r) * N + n0 + cc], 0, 1);
      v8bf v = ld8(&W[(size_t)(k0 + r) * N + n0 + cc]);
#pragma unroll
      for (int j = 0; j < 8; ++j) Wt[(cc + j) * 40 + r] = v[j];
    }
    if (wave == 0) __builtin_amdgcn_s_wait_tensorcnt(0);
    __syncthreads();
    v16bf af[2], bf[2];
#pragma unroll
    for (int mt = 0; mt < 2; ++mt) {
      int m = wm * 32 + mt * 16 + ln;
      int off = half * 16;
      af[mt] = mk16(ld8(&As[m * 40 + off]), ld8(&As[m * 40 + off + 8]));
    }
#pragma unroll
    for (int nt = 0; nt < 2; ++nt) {
      int n = wn * 32 + nt * 16 + ln;
      int base = half * 16;
      bf[nt] = mk16(ld8(&Wt[n * 40 + base]), ld8(&Wt[n * 40 + base + 8]));
    }
#pragma unroll
    for (int mt = 0; mt < 2; ++mt)
#pragma unroll
      for (int nt = 0; nt < 2; ++nt)
        acc[mt][nt] = WMMA_BF16(af[mt], bf[nt], acc[mt][nt]);
    __syncthreads();
  }
#pragma unroll
  for (int mt = 0; mt < 2; ++mt)
#pragma unroll
    for (int nt = 0; nt < 2; ++nt) {
      int n = n0 + wn * 32 + nt * 16 + ln;
      float bv = bias[n];
#pragma unroll
      for (int e = 0; e < 8; ++e) {
        int m = m0 + wm * 32 + mt * 16 + e + half * 8;
        C[(size_t)m * N + n] = (__bf16)(acc[mt][nt][e] + bv);
      }
    }
}

// ---------------- flash attention, accumulating sum_q(ctx) into ctxsum[b, h*96+d] ------
// K tile staged by TDM (row-major, stride 96); V tile manually transposed for B-fragments.
__global__ __launch_bounds__(128) void k_flash(
    const __bf16* __restrict__ Q, const __bf16* __restrict__ K,
    const __bf16* __restrict__ V, const int* __restrict__ mask,
    float* __restrict__ ctxsum) {
  const int S = 1024, D = 768, HD = 96;
  const float scale = 0.10206207261596575f;  // 96^-0.5
  __shared__ __align__(16) __bf16 Ks[64 * 96];
  __shared__ __align__(16) __bf16 Vt[96 * 72];
  __shared__ __align__(16) __bf16 Ps[4 * 16 * 64];
  int tid = threadIdx.x, lane = tid & 31, wave = tid >> 5;
  int ln = lane & 15, half = lane >> 4;
  int qb = blockIdx.x, h = blockIdx.y, b = blockIdx.z;
  const size_t base = (size_t)b * S * D + h * HD;

  // Q fragments for this wave's 16 q rows (A-matrix 16x32 layout, 3 k-steps over hd=96)
  int qrow = qb * 64 + wave * 16 + ln;
  v16bf qf[3];
#pragma unroll
  for (int ks = 0; ks < 3; ++ks) {
    const __bf16* qp = Q + base + (size_t)qrow * D + ks * 32;
    int off = half * 16;
    qf[ks] = mk16(ld8(qp + off), ld8(qp + off + 8));
  }

  float m[8], l[8];
  v8f cacc[6];
#pragma unroll
  for (int e = 0; e < 8; ++e) { m[e] = -3.0e38f; l[e] = 0.f; }
#pragma unroll
  for (int c = 0; c < 6; ++c)
#pragma unroll
    for (int e = 0; e < 8; ++e) cacc[c][e] = 0.f;

  __bf16* Pw = &Ps[wave * 16 * 64];

  for (int kt = 0; kt < 16; ++kt) {
    __syncthreads();
    // K tile 64x96 row-major via TDM
    if (wave == 0) {
      tdm_load_2d((unsigned int)(size_t)Ks, K + base + (size_t)(kt * 64) * D,
                  /*dim0=*/96, /*dim1=*/64, /*stride0=*/(unsigned int)D, 0u);
    }
    // V tile transposed -> Vt[d][key] (manual: TDM cannot transpose)
#pragma unroll
    for (int i = 0; i < 6; ++i) {
      int c = tid + i * 128;
      int r = c / 12, cc = (c % 12) * 8;
      v8bf vv = ld8(&V[base + (size_t)(kt * 64 + r) * D + cc]);
#pragma unroll
      for (int j = 0; j < 8; ++j) Vt[(cc + j) * 72 + r] = vv[j];
    }
    if (wave == 0) __builtin_amdgcn_s_wait_tensorcnt(0);
    __syncthreads();

    // scores for 64 keys (4 N-tiles x 3 k-steps of WMMA)
    v8f sc[4];
#pragma unroll
    for (int nt = 0; nt < 4; ++nt) {
      v8f a;
#pragma unroll
      for (int e = 0; e < 8; ++e) a[e] = 0.f;
      int key = nt * 16 + ln;
#pragma unroll
      for (int ks = 0; ks < 3; ++ks) {
        int bb = ks * 32 + half * 16;
        v16bf bfr = mk16(ld8(&Ks[key * 96 + bb]), ld8(&Ks[key * 96 + bb + 8]));
        a = WMMA_BF16(qf[ks], bfr, a);
      }
      int kg = kt * 64 + nt * 16 + ln;
      float mb = (mask[b * S + kg] == 0) ? -3.0e38f : 0.0f;
#pragma unroll
      for (int e = 0; e < 8; ++e) a[e] = a[e] * scale + mb;
      sc[nt] = a;
    }
    // row max (reduce over 16 lanes sharing a row)
    float mn[8], sfac[8];
#pragma unroll
    for (int e = 0; e < 8; ++e) {
      float t = fmaxf(fmaxf(sc[0][e], sc[1][e]), fmaxf(sc[2][e], sc[3][e]));
#pragma unroll
      for (int o = 1; o < 16; o <<= 1) t = fmaxf(t, __shfl_xor(t, o, 16));
      mn[e] = fmaxf(m[e], t);
      sfac[e] = expf(m[e] - mn[e]);
      m[e] = mn[e];
    }
    // P = exp(sc - m), row sum, stage P to LDS (per-wave region) as bf16
    float rs[8];
#pragma unroll
    for (int e = 0; e < 8; ++e) rs[e] = 0.f;
#pragma unroll
    for (int nt = 0; nt < 4; ++nt)
#pragma unroll
      for (int e = 0; e < 8; ++e) {
        float p = expf(sc[nt][e] - mn[e]);
        rs[e] += p;
        Pw[(e + half * 8) * 64 + nt * 16 + ln] = (__bf16)p;
      }
#pragma unroll
    for (int e = 0; e < 8; ++e) {
#pragma unroll
      for (int o = 1; o < 16; o <<= 1) rs[e] += __shfl_xor(rs[e], o, 16);
      l[e] = l[e] * sfac[e] + rs[e];
    }
#pragma unroll
    for (int c = 0; c < 6; ++c)
#pragma unroll
      for (int e = 0; e < 8; ++e) cacc[c][e] *= sfac[e];
    // re-layout P into A fragments, ctx += P @ V
    v16bf pf[2];
#pragma unroll
    for (int pk = 0; pk < 2; ++pk) {
      int off = half * 16;
      pf[pk] = mk16(ld8(&Pw[ln * 64 + pk * 32 + off]), ld8(&Pw[ln * 64 + pk * 32 + off + 8]));
    }
#pragma unroll
    for (int c = 0; c < 6; ++c) {
      int vd = c * 16 + ln;
#pragma unroll
      for (int pk = 0; pk < 2; ++pk) {
        int bb = pk * 32 + half * 16;
        v16bf bfr = mk16(ld8(&Vt[vd * 72 + bb]), ld8(&Vt[vd * 72 + bb + 8]));
        cacc[c] = WMMA_BF16(pf[pk], bfr, cacc[c]);
      }
    }
  }
  // normalize rows by 1/l, sum over the 16 q rows, accumulate to global
  float inv[8];
#pragma unroll
  for (int e = 0; e < 8; ++e) inv[e] = 1.0f / l[e];
#pragma unroll
  for (int c = 0; c < 6; ++c) {
    float p = 0.f;
#pragma unroll
    for (int e = 0; e < 8; ++e) p += cacc[c][e] * inv[e];
    p += __shfl_xor(p, 16);
    if (half == 0) atomicAdd(&ctxsum[b * 768 + h * 96 + c * 16 + ln], p);
  }
}

// ---------------- pooled repr: sum_s w[b,s]*h[b,s,d] for both branches ----------------
__global__ void k_pool(const float* __restrict__ h, const float* __restrict__ aw,
                       const float* __restrict__ ow, float* __restrict__ ap,
                       float* __restrict__ op) {
  int d = blockIdx.x * 256 + threadIdx.x;
  int b = blockIdx.y;
  float sa = 0.f, so = 0.f;
  for (int s = 0; s < 1024; ++s) {
    float hv = h[((size_t)b * 1024 + s) * 768 + d];
    sa += aw[b * 1024 + s] * hv;
    so += ow[b * 1024 + s] * hv;
  }
  ap[b * 768 + d] = sa;
  op[b * 768 + d] = so;
}

// repr = pool + (ctxsum/S) @ out_w + out_b   (mean pushed through linear out-proj)
__global__ void k_repr(const float* __restrict__ pool, const float* __restrict__ ctx,
                       const float* __restrict__ outw, const float* __restrict__ outb,
                       float* __restrict__ repr) {
  int d = blockIdx.x * 256 + threadIdx.x;
  int b = blockIdx.y;
  const float invS = 1.0f / 1024.0f;
  float acc = pool[b * 768 + d] + outb[d];
  for (int k = 0; k < 768; ++k) acc += ctx[b * 768 + k] * invS * outw[k * 768 + d];
  repr[b * 768 + d] = acc;
}

__global__ void k_gate(const float* __restrict__ ar, const float* __restrict__ orp,
                       const float* __restrict__ gw, const float* __restrict__ gb,
                       float* __restrict__ fused) {
  int d = blockIdx.x * 256 + threadIdx.x;
  int b = blockIdx.y;
  float g = gb[d];
  for (int k = 0; k < 768; ++k) g += ar[b * 768 + k] * gw[k * 768 + d];
  for (int k = 0; k < 768; ++k) g += orp[b * 768 + k] * gw[(768 + k) * 768 + d];
  g = 1.0f / (1.0f + expf(-g));
  fused[b * 768 + d] = g * ar[b * 768 + d] + (1.0f - g) * orp[b * 768 + d];
}

__global__ void k_fusion(const float* __restrict__ fused, const float* __restrict__ ar,
                         const float* __restrict__ orp, const float* __restrict__ fw,
                         const float* __restrict__ fb, float* __restrict__ fin) {
  int d = blockIdx.x * 256 + threadIdx.x;
  int b = blockIdx.y;
  float a = fb[d];
  for (int k = 0; k < 768; ++k) a += fused[b * 768 + k] * fw[k * 768 + d];
  for (int k = 0; k < 768; ++k) a += ar[b * 768 + k] * fw[(768 + k) * 768 + d];
  for (int k = 0; k < 768; ++k) a += orp[b * 768 + k] * fw[(1536 + k) * 768 + d];
  fin[b * 768 + d] = geluf(a);
}

__global__ __launch_bounds__(256) void k_head(
    const float* __restrict__ fin,
    const float* __restrict__ c1w, const float* __restrict__ c1b,
    const float* __restrict__ c2w, const float* __restrict__ c2b,
    const float* __restrict__ f1w, const float* __restrict__ f1b,
    const float* __restrict__ f2w, const float* __restrict__ f2b,
    float* __restrict__ out) {
  __shared__ float shx[768], sh1[768], shc[384], red[256];
  int b = blockIdx.x, tid = threadIdx.x;
  for (int j = tid; j < 768; j += 256) shx[j] = fin[b * 768 + j];
  __syncthreads();
  for (int j = tid; j < 768; j += 256) {
    float a = c1b[j];
    for (int k = 0; k < 768; ++k) a += shx[k] * c1w[k * 768 + j];
    sh1[j] = geluf(a);
  }
  for (int j = tid; j < 384; j += 256) {
    float a = f1b[j];
    for (int k = 0; k < 768; ++k) a += shx[k] * f1w[k * 384 + j];
    shc[j] = geluf(a);
  }
  __syncthreads();
  if (tid < 3) {
    float a = c2b[tid];
    for (int k = 0; k < 768; ++k) a += sh1[k] * c2w[k * 3 + tid];
    out[b * 3 + tid] = a;
  }
  float p = 0.f;
  for (int k = tid; k < 384; k += 256) p += shc[k] * f2w[k];
  red[tid] = p; __syncthreads();
  for (int o = 128; o > 0; o >>= 1) { if (tid < o) red[tid] += red[tid + o]; __syncthreads(); }
  if (tid == 0) out[24 + b] = 1.0f / (1.0f + expf(-(red[0] + f2b[0])));
}

extern "C" void kernel_launch(void* const* d_in, const int* in_sizes, int n_in,
                              void* d_out, int out_size, void* d_ws, size_t ws_size,
                              hipStream_t stream) {
  (void)in_sizes; (void)n_in; (void)out_size; (void)ws_size;
  const float* hidden = (const float*)d_in[0];
  const float* alog = (const float*)d_in[1];
  const float* olog = (const float*)d_in[2];
  // d_in[3] span_features: unused by the reference forward
  const int* mask = (const int*)d_in[4];
  const float* aqw = (const float*)d_in[5];  const float* aqb = (const float*)d_in[6];
  const float* akw = (const float*)d_in[7];  const float* akb = (const float*)d_in[8];
  const float* avw = (const float*)d_in[9];  const float* avb = (const float*)d_in[10];
  const float* aow = (const float*)d_in[11]; const float* aob = (const float*)d_in[12];
  const float* oqw = (const float*)d_in[13]; const float* oqb = (const float*)d_in[14];
  const float* okw = (const float*)d_in[15]; const float* okb = (const float*)d_in[16];
  const float* ovw = (const float*)d_in[17]; const float* ovb = (const float*)d_in[18];
  const float* oow = (const float*)d_in[19]; const float* oob = (const float*)d_in[20];
  // d_in[21..28] context_* : dead code under DCE
  const float* gw  = (const float*)d_in[29]; const float* gb  = (const float*)d_in[30];
  const float* fw  = (const float*)d_in[31]; const float* fb  = (const float*)d_in[32];
  const float* c1w = (const float*)d_in[33]; const float* c1b = (const float*)d_in[34];
  const float* c2w = (const float*)d_in[35]; const float* c2b = (const float*)d_in[36];
  const float* f1w = (const float*)d_in[37]; const float* f1b = (const float*)d_in[38];
  const float* f2w = (const float*)d_in[39]; const float* f2b = (const float*)d_in[40];

  const int B = 8, S = 1024, D = 768;
  const size_t BSD = (size_t)B * S * D;

  char* wsp = (char*)d_ws;
  size_t off = 0;
  auto alloc = [&](size_t bytes) -> void* {
    void* p = wsp + off;
    off = (off + bytes + 255) & ~(size_t)255;
    return p;
  };
  __bf16* hb = (__bf16*)alloc(BSD * 2);
  __bf16* wQ = (__bf16*)alloc((size_t)D * D * 2);
  __bf16* wK = (__bf16*)alloc((size_t)D * D * 2);
  __bf16* wV = (__bf16*)alloc((size_t)D * D * 2);
  __bf16* Qb = (__bf16*)alloc(BSD * 2);
  __bf16* Kb = (__bf16*)alloc(BSD * 2);
  __bf16* Vb = (__bf16*)alloc(BSD * 2);
  float* awb = (float*)alloc((size_t)B * S * 4);
  float* owb = (float*)alloc((size_t)B * S * 4);
  float* ctxA = (float*)alloc((size_t)B * D * 4);
  float* ctxO = (float*)alloc((size_t)B * D * 4);
  float* apool = (float*)alloc((size_t)B * D * 4);
  float* opool = (float*)alloc((size_t)B * D * 4);
  float* arepr = (float*)alloc((size_t)B * D * 4);
  float* orepr = (float*)alloc((size_t)B * D * 4);
  float* fusedb = (float*)alloc((size_t)B * D * 4);
  float* finalb = (float*)alloc((size_t)B * D * 4);

  hipMemsetAsync(ctxA, 0, (size_t)B * D * 4, stream);
  hipMemsetAsync(ctxO, 0, (size_t)B * D * 4, stream);

  k_cvt_bf16<<<1024, 256, 0, stream>>>(hidden, hb, (int)BSD);
  k_tok_weights<<<8, 256, 0, stream>>>(alog, olog, awb, owb);

  dim3 ggrid(D / 64, (B * S) / 128);  // (12, 64)
  dim3 fgrid(S / 64, 8, B);           // (16, 8, 8)

  // aspect branch
  k_cvt_bf16<<<256, 256, 0, stream>>>(aqw, wQ, D * D);
  k_cvt_bf16<<<256, 256, 0, stream>>>(akw, wK, D * D);
  k_cvt_bf16<<<256, 256, 0, stream>>>(avw, wV, D * D);
  k_gemm_bf16<<<ggrid, 256, 0, stream>>>(hb, wQ, aqb, Qb, B * S, D, D);
  k_gemm_bf16<<<ggrid, 256, 0, stream>>>(hb, wK, akb, Kb, B * S, D, D);
  k_gemm_bf16<<<ggrid, 256, 0, stream>>>(hb, wV, avb, Vb, B * S, D, D);
  k_flash<<<fgrid, 128, 0, stream>>>(Qb, Kb, Vb, mask, ctxA);

  // opinion branch (reuses QKV/weight buffers; stream-ordered)
  k_cvt_bf16<<<256, 256, 0, stream>>>(oqw, wQ, D * D);
  k_cvt_bf16<<<256, 256, 0, stream>>>(okw, wK, D * D);
  k_cvt_bf16<<<256, 256, 0, stream>>>(ovw, wV, D * D);
  k_gemm_bf16<<<ggrid, 256, 0, stream>>>(hb, wQ, oqb, Qb, B * S, D, D);
  k_gemm_bf16<<<ggrid, 256, 0, stream>>>(hb, wK, okb, Kb, B * S, D, D);
  k_gemm_bf16<<<ggrid, 256, 0, stream>>>(hb, wV, ovb, Vb, B * S, D, D);
  k_flash<<<fgrid, 128, 0, stream>>>(Qb, Kb, Vb, mask, ctxO);

  // small tail
  k_pool<<<dim3(3, 8), 256, 0, stream>>>(hidden, awb, owb, apool, opool);
  k_repr<<<dim3(3, 8), 256, 0, stream>>>(apool, ctxA, aow, aob, arepr);
  k_repr<<<dim3(3, 8), 256, 0, stream>>>(opool, ctxO, oow, oob, orepr);
  k_gate<<<dim3(3, 8), 256, 0, stream>>>(arepr, orepr, gw, gb, fusedb);
  k_fusion<<<dim3(3, 8), 256, 0, stream>>>(fusedb, arepr, orepr, fw, fb, finalb);
  k_head<<<8, 256, 0, stream>>>(finalb, c1w, c1b, c2w, c2b, f1w, f1b, f2w, f2b,
                                (float*)d_out);
}
